// cheb_conv_orignal_32512902431520
// MI455X (gfx1250) — matrix operand order repaired and checked by reading, not compiled, not run on hardware
//
#include <hip/hip_runtime.h>
#include <cstddef>

// Problem constants (from reference): K=3, N=1024, C_IN=32, C_OUT=64, B=32, T=12
// out[b,n,f,t] = relu( sum_{k,c,m} cheb[k,m,n] * x[b,m,c,t] * Theta[k,c,f] )
//
// Strategy (MI455X / gfx1250, wave32):
//  1) one-shot f32->f16 conversion kernels that also pre-transpose:
//       chebT[k][n][m]              (A operand, row-major in m)
//       xT[b][j'][m], j'=t*32+c     (B operand, row-major in m; t-major cols)
//       ThetaT[k][f][c]             (B operand of epilogue WMMA)
//  2) fused GEMM kernel: per (b, 32-row n-tile) workgroup of 8 waves,
//     each wave: 16 n-rows x 96 j'-cols x 3 k  f32 accumulators (144 VGPRs),
//     main loop over m in chunks of 32 using v_wmma_f32_16x16x32_f16.
//     Since 96 t-major columns == 3 full t-slices (all 32 c), the Theta
//     contraction is wave-local: R tile -> LDS -> A-fragment -> second WMMA
//     against ThetaT, ReLU, store. Each output written exactly once.
//     Epilogue is FULLY unrolled so the accumulator array is never
//     runtime-indexed (avoids scratch demotion / spills).
//
// Workspace: 6 MiB (chebT) + 24 MiB (xT) + 12 KiB (ThetaT) ~= 31.5 MB.

typedef __attribute__((ext_vector_type(16))) _Float16 v16h;
typedef __attribute__((ext_vector_type(8)))  _Float16 v8h;
typedef __attribute__((ext_vector_type(8)))  float    v8f;

union V16 { v16h v; v8h h[2]; };

#define GN    1024           // graph nodes
#define GK    3              // cheb order
#define GCIN  32
#define GCOUT 64
#define GB    32
#define GT    12
#define GJ    (GT * GCIN)    // 384 columns per batch

// ---------------- conversion kernels (f32 -> f16, with transpose) ------------

__global__ void convert_cheb(const float* __restrict__ cheb,
                             _Float16* __restrict__ chebT) {
  int i = blockIdx.x * blockDim.x + threadIdx.x;       // over GK*GN*GN
  if (i >= GK * GN * GN) return;
  int k = i >> 20;                 // /(GN*GN)
  int r = i & (GN * GN - 1);
  int n = r >> 10;
  int m = r & (GN - 1);
  // chebT[k][n][m] = cheb[k][m][n]
  chebT[i] = (_Float16)cheb[(size_t)k * GN * GN + (size_t)m * GN + n];
}

__global__ void convert_x(const float* __restrict__ x,
                          _Float16* __restrict__ xT) {
  int i = blockIdx.x * blockDim.x + threadIdx.x;       // over GB*GJ*GN
  if (i >= GB * GJ * GN) return;
  int m = i & (GN - 1);
  int r = i >> 10;
  int j = r % GJ;                  // j' = t*32 + c  (t-major!)
  int b = r / GJ;
  int t = j >> 5;
  int c = j & 31;
  // xT[b][j'][m] = x[b][m][c][t]
  xT[i] = (_Float16)x[(size_t)b * GN * GCIN * GT + (size_t)m * GCIN * GT
                      + (size_t)c * GT + t];
}

__global__ void convert_theta(const float* __restrict__ theta,
                              _Float16* __restrict__ thT) {
  int i = blockIdx.x * blockDim.x + threadIdx.x;       // over GK*GCOUT*GCIN
  if (i >= GK * GCOUT * GCIN) return;
  int c = i & 31;
  int r = i >> 5;
  int f = r % GCOUT;
  int k = r / GCOUT;
  // ThetaT[k][f][c] = Theta[k][c][f]
  thT[i] = (_Float16)theta[(size_t)k * GCIN * GCOUT + (size_t)c * GCOUT + f];
}

// ---------------- fused WMMA GEMM + Theta contraction + ReLU -----------------

__global__ __launch_bounds__(256, 1) void
cheb_gemm(const _Float16* __restrict__ chebT,
          const _Float16* __restrict__ xT,
          const _Float16* __restrict__ thT,
          float* __restrict__ out) {
  // per-wave private LDS re-layout buffer: 8 waves x 3 k x (16 rows x 32 c) f16
  __shared__ __align__(16) _Float16 lds[8][GK][16 * 32];

  const int lane = threadIdx.x & 31;
  const int w    = threadIdx.x >> 5;     // wave 0..7
  const int rg   = w >> 2;               // row group 0..1 (16 n-rows each)
  const int cg   = w & 3;                // col group 0..3 (96 j'-cols each)
  const int b    = blockIdx.y;
  const int n0   = blockIdx.x * 32 + rg * 16;   // wave's first n row
  const int hi   = (lane & 16) ? 1 : 0;         // lane half
  const int row  = lane & 15;

  const _Float16* __restrict__ xb = xT + (size_t)b * GJ * GN;

  v8f acc[GK][6];
#pragma unroll
  for (int k = 0; k < GK; ++k)
#pragma unroll
    for (int jt = 0; jt < 6; ++jt)
      acc[k][jt] = (v8f){0.f, 0.f, 0.f, 0.f, 0.f, 0.f, 0.f, 0.f};

  // ---- main GEMM loop over m in chunks of 32 (WMMA K) ----
#pragma unroll 1
  for (int m0 = 0; m0 < GN; m0 += 32) {
    // A fragments: chebT[k][n0+row][m0 + ...]
    // layout (16-bit A 16x32): halves 0..7 = K kb..kb+7, 8..15 = K kb+16..kb+23
    V16 aF[GK];
#pragma unroll
    for (int k = 0; k < GK; ++k) {
      const _Float16* p = chebT + ((size_t)k << 20)
                        + (size_t)(n0 + row) * GN + m0 + hi * 8;
      aF[k].h[0] = *(const v8h*)(p);
      aF[k].h[1] = *(const v8h*)(p + 16);
    }
#pragma unroll
    for (int jt = 0; jt < 6; ++jt) {
      // B fragment: column j' = cg*96 + jt*16 + row; K (=m) kb..kb+7, kb+16..kb+23
      const _Float16* p = xb + (size_t)(cg * 96 + jt * 16 + row) * GN
                        + m0 + hi * 8;
      V16 bF;
      bF.h[0] = *(const v8h*)(p);
      bF.h[1] = *(const v8h*)(p + 16);
#pragma unroll
      for (int k = 0; k < GK; ++k) {
        acc[k][jt] = __builtin_amdgcn_wmma_f32_16x16x32_f16(
            false, aF[k].v, false, bF.v, (short)0, acc[k][jt], false, false);
      }
    }
  }

  // ---- epilogue: per t-slice, contract with Theta via a 2nd WMMA chain ----
  // wave's 96 cols = t in {cg*3, cg*3+1, cg*3+2}, all 32 c per t (t-major xT).
  // FULLY unrolled: all acc[][] indices are compile-time constants so the
  // accumulator array stays in VGPRs and dead tiles free registers per slice.
#pragma unroll
  for (int tl = 0; tl < 3; ++tl) {
    // store R_k tile (16 rows x 32 c) into wave-private LDS, f32 -> f16
#pragma unroll
    for (int k = 0; k < GK; ++k) {
#pragma unroll
      for (int j2 = 0; j2 < 2; ++j2) {
        const int jt = tl * 2 + j2;
#pragma unroll
        for (int v = 0; v < 8; ++v) {
          const int r2 = v + hi * 8;                 // local n row 0..15
          lds[w][k][r2 * 32 + j2 * 16 + row] = (_Float16)acc[k][jt][v];
        }
      }
    }
    __syncthreads();   // uniform; orders wave's ds stores before ds loads

    v8f d[4];
#pragma unroll
    for (int ft = 0; ft < 4; ++ft)
      d[ft] = (v8f){0.f, 0.f, 0.f, 0.f, 0.f, 0.f, 0.f, 0.f};

#pragma unroll
    for (int k = 0; k < GK; ++k) {
      // A fragment from LDS: rows = local n, K = c (32)
      V16 aF;
      const _Float16* p = &lds[w][k][row * 32 + hi * 8];
      aF.h[0] = *(const v8h*)(p);
      aF.h[1] = *(const v8h*)(p + 16);
#pragma unroll
      for (int ft = 0; ft < 4; ++ft) {
        // B fragment: ThetaT[k][f = ft*16+row][c ...], c = hi*8+0..7, 16+hi*8+0..7
        const _Float16* q = thT + (size_t)k * GCOUT * GCIN
                          + (size_t)(ft * 16 + row) * GCIN + hi * 8;
        V16 bF;
        bF.h[0] = *(const v8h*)(q);
        bF.h[1] = *(const v8h*)(q + 16);
        d[ft] = __builtin_amdgcn_wmma_f32_16x16x32_f16(
            false, aF.v, false, bF.v, (short)0, d[ft], false, false);
      }
    }
    __syncthreads();

    // ReLU + store: out[b][n][f][t], each element written exactly once
    const int t = cg * 3 + tl;
#pragma unroll
    for (int ft = 0; ft < 4; ++ft) {
      const int f = ft * 16 + row;
#pragma unroll
      for (int v = 0; v < 8; ++v) {
        const int n = n0 + v + hi * 8;
        const float val = d[ft][v];
        out[(((size_t)b * GN + n) * GCOUT + f) * GT + t] = val > 0.f ? val : 0.f;
      }
    }
  }
}

// ---------------- launch -----------------------------------------------------

extern "C" void kernel_launch(void* const* d_in, const int* in_sizes, int n_in,
                              void* d_out, int out_size, void* d_ws, size_t ws_size,
                              hipStream_t stream) {
  const float* x     = (const float*)d_in[0];   // [B,N,C_IN,T]
  const float* cheb  = (const float*)d_in[1];   // [K,N,N]
  const float* theta = (const float*)d_in[2];   // [K,C_IN,C_OUT]
  float* out = (float*)d_out;                   // [B,N,C_OUT,T]

  // workspace layout (f16): chebT | xT | ThetaT  (~31.5 MB total)
  const size_t chebT_bytes = (size_t)GK * GN * GN * sizeof(_Float16);      // 6 MiB
  const size_t xT_bytes    = (size_t)GB * GJ * GN * sizeof(_Float16);      // 24 MiB
  _Float16* chebT = (_Float16*)d_ws;
  _Float16* xT    = (_Float16*)((char*)d_ws + chebT_bytes);
  _Float16* thT   = (_Float16*)((char*)d_ws + chebT_bytes + xT_bytes);

  convert_cheb<<<(GK * GN * GN + 255) / 256, 256, 0, stream>>>(cheb, chebT);
  convert_x<<<(GB * GJ * GN + 255) / 256, 256, 0, stream>>>(x, xT);
  convert_theta<<<(GK * GCOUT * GCIN + 255) / 256, 256, 0, stream>>>(theta, thT);

  cheb_gemm<<<dim3(GN / 32, GB), 256, 0, stream>>>(chebT, xT, thT, out);
}
